// GraphPFN_59390807769610
// MI455X (gfx1250) — compile-verified, matrix operand order checked by reading, not run on hardware
//
#include <hip/hip_runtime.h>
#include <hip/hip_bf16.h>
#include <math.h>

#define NNODES 100000
#define NEDGES 1600000
#define DMODEL 192
#define NHEAD  4
#define DHEAD  48

typedef __attribute__((ext_vector_type(16))) _Float16 v16h;
typedef __attribute__((ext_vector_type(8)))  _Float16 v8h;
typedef __attribute__((ext_vector_type(8)))  float    v8f;

// ---------------- permutation (stable binary partition) ----------------
// pos[j]  = sorted position of node j (train nodes first, stable)
// invp[p] = node at sorted position p
__global__ void gpfn_perm_kernel(const unsigned char* __restrict__ mask,
                                 int* __restrict__ pos, int* __restrict__ invp) {
    __shared__ int sdata[1024];
    __shared__ int sTotalTrain;
    __shared__ int sOffT, sOffF;
    const int tid = threadIdx.x;

    // phase 1: total train count
    int cnt = 0;
    for (int j = tid; j < NNODES; j += 1024) cnt += (mask[j] ? 1 : 0);
    sdata[tid] = cnt; __syncthreads();
    for (int s = 512; s > 0; s >>= 1) {
        if (tid < s) sdata[tid] += sdata[tid + s];
        __syncthreads();
    }
    if (tid == 0) { sTotalTrain = sdata[0]; sOffT = 0; sOffF = 0; }
    __syncthreads();
    const int totalTrain = sTotalTrain;

    // phase 2: chunked stable scan
    for (int base = 0; base < NNODES; base += 1024) {
        const int j = base + tid;
        const int valid = (j < NNODES) ? 1 : 0;
        const int isT = (valid && mask[j]) ? 1 : 0;
        sdata[tid] = isT; __syncthreads();
        for (int s2 = 1; s2 < 1024; s2 <<= 1) {
            int v = (tid >= s2) ? sdata[tid - s2] : 0;
            __syncthreads();
            sdata[tid] += v;
            __syncthreads();
        }
        const int incT = sdata[tid];
        const int exclT = incT - isT;
        const int chunkTrain = sdata[1023];
        const int rem = NNODES - base;
        const int chunkValid = (rem < 1024) ? rem : 1024;
        const int offT = sOffT, offF = sOffF;
        if (valid) {
            int p;
            if (isT) p = offT + exclT;
            else     p = totalTrain + offF + (tid - exclT);
            pos[j] = p;
            invp[p] = j;
        }
        __syncthreads();
        if (tid == 0) { sOffT += chunkTrain; sOffF += chunkValid - chunkTrain; }
        __syncthreads();
    }
}

// ---------------- weight convert + transpose: W[K][Nc] f32 -> Wt[Nc][K] f16 ----
__global__ void gpfn_convert_wt_kernel(const float* __restrict__ W,
                                       _Float16* __restrict__ Wt, int K, int Nc) {
    int idx = blockIdx.x * blockDim.x + threadIdx.x;
    if (idx >= K * Nc) return;
    int k = idx / Nc, n = idx % Nc;
    Wt[(size_t)n * K + k] = (_Float16)W[idx];
}

// ---------------- gather + layernorm (one wave per row) ----------------
template <typename OUT_T>
__global__ void gpfn_gather_ln_kernel(const float* __restrict__ xin,
                                      const int* __restrict__ gidx,
                                      const float* __restrict__ g,
                                      const float* __restrict__ b,
                                      float* __restrict__ hcopy,
                                      OUT_T* __restrict__ hnorm) {
    const int row  = (int)((blockIdx.x * blockDim.x + threadIdx.x) >> 5);
    const int lane = threadIdx.x & 31;
    if (row >= NNODES) return;
    const float* src = xin + (size_t)gidx[row] * DMODEL;
    float vals[6];
    float s = 0.f;
#pragma unroll
    for (int i = 0; i < 6; ++i) { vals[i] = src[lane + 32 * i]; s += vals[i]; }
#pragma unroll
    for (int off = 16; off > 0; off >>= 1) s += __shfl_xor(s, off, 32);
    const float mu = s * (1.0f / DMODEL);
    float vs = 0.f;
#pragma unroll
    for (int i = 0; i < 6; ++i) { float d = vals[i] - mu; vs += d * d; }
#pragma unroll
    for (int off = 16; off > 0; off >>= 1) vs += __shfl_xor(vs, off, 32);
    const float inv = rsqrtf(vs * (1.0f / DMODEL) + 1e-5f);
    float* hc = hcopy + (size_t)row * DMODEL;
    OUT_T* hn = hnorm + (size_t)row * DMODEL;
#pragma unroll
    for (int i = 0; i < 6; ++i) {
        const int c = lane + 32 * i;
        hc[c] = vals[i];
        hn[c] = (OUT_T)((vals[i] - mu) * inv * g[c] + b[c]);
    }
}

// ---------------- WMMA GEMM: C[M][NDIM] = A[M][KDIM] @ Bt'[KDIM][NDIM] ----------
// Bt stored transposed: Bt[n][k], f16. One wave per 16x64 output strip
// (register-blocked over N: A fragment loaded once per k-step, 4 WMMAs).
// EPI: 0 = +bias ; 1 = +bias +res ; 2 = gelu(+bias)
template <typename AT, int KDIM, int NDIM, int EPI>
__global__ void gpfn_wmma_gemm_kernel(const AT* __restrict__ A,
                                      const _Float16* __restrict__ Bt,
                                      const float* __restrict__ bias,
                                      const float* __restrict__ res,
                                      float* __restrict__ C, int Mrows) {
    constexpr int NT4 = NDIM / 64;          // 16x64 strips along N
    const int wave = (int)((blockIdx.x * blockDim.x + threadIdx.x) >> 5);
    const int lane = threadIdx.x & 31;
    const int mt = wave / NT4, nt = wave % NT4;
    if (mt * 16 >= Mrows) return;
    const int row0 = mt * 16, col0 = nt * 64;
    const int mn  = lane & 15;
    const int hi8 = (lane & 16) ? 8 : 0;

    const AT* arow = A + (size_t)(row0 + mn) * KDIM;
    const _Float16* bbase = Bt + (size_t)(col0 + mn) * KDIM;

    v8f acc[4] = {};
    for (int k0 = 0; k0 < KDIM; k0 += 32) {
        v16h a;
        if (sizeof(AT) == 2) {
            v8h alo = *(const v8h*)(const void*)(arow + k0 + hi8);
            v8h ahi = *(const v8h*)(const void*)(arow + k0 + 16 + hi8);
#pragma unroll
            for (int e = 0; e < 8; ++e) { a[e] = (_Float16)alo[e]; a[e + 8] = (_Float16)ahi[e]; }
        } else {
#pragma unroll
            for (int e = 0; e < 8; ++e) {
                a[e]     = (_Float16)arow[k0 + hi8 + e];
                a[e + 8] = (_Float16)arow[k0 + 16 + hi8 + e];
            }
        }
#pragma unroll
        for (int j = 0; j < 4; ++j) {
            const _Float16* bcol = bbase + (size_t)(j * 16) * KDIM;
            v8h blo = *(const v8h*)(const void*)(bcol + k0 + hi8);
            v8h bhi = *(const v8h*)(const void*)(bcol + k0 + 16 + hi8);
            v16h b;
#pragma unroll
            for (int e = 0; e < 8; ++e) { b[e] = blo[e]; b[e + 8] = bhi[e]; }
            acc[j] = __builtin_amdgcn_wmma_f32_16x16x32_f16(
                /*neg_a=*/false, a, /*neg_b=*/false, b,
                /*c_mod=*/(short)0, acc[j], /*reuse_a=*/false, /*reuse_b=*/false);
        }
    }

#pragma unroll
    for (int j = 0; j < 4; ++j) {
        const int n = col0 + j * 16 + mn;
        const float bn = bias[n];
#pragma unroll
        for (int r = 0; r < 8; ++r) {
            const int m = row0 + r + hi8;
            float v = acc[j][r] + bn;
            if (EPI == 1) v += res[(size_t)m * NDIM + n];
            if (EPI == 2) v = 0.5f * v * (1.0f + erff(v * 0.70710678118654752f));
            C[(size_t)m * NDIM + n] = v;
        }
    }
}

// ---------------- edge softmax phase ----------------
__device__ __forceinline__ unsigned gpfn_encodeF(float f) {
    unsigned u = __float_as_uint(f);
    return (u & 0x80000000u) ? ~u : (u | 0x80000000u);
}
__device__ __forceinline__ float gpfn_decodeF(unsigned u) {
    u = (u & 0x80000000u) ? (u & 0x7FFFFFFFu) : ~u;
    return __uint_as_float(u);
}

__global__ void gpfn_init_m_kernel(unsigned* __restrict__ mEnc, int n) {
    int i = blockIdx.x * blockDim.x + threadIdx.x;
    if (i < n) mEnc[i] = 0x007FFFFFu; // encode(-inf)
}

__global__ void gpfn_edge_score_kernel(const int* __restrict__ src,
                                       const int* __restrict__ dst,
                                       const float* __restrict__ qkv,
                                       float* __restrict__ score,
                                       unsigned* __restrict__ mEnc) {
    int t = blockIdx.x * blockDim.x + threadIdx.x;
    if (t >= NEDGES * NHEAD) return;
    const int e = t >> 2, h = t & 3;
    const int s = src[e], d = dst[e];
    const float* kp = qkv + (size_t)s * (3 * DMODEL) + h * (3 * DHEAD) + DHEAD;
    const float* qp = qkv + (size_t)d * (3 * DMODEL) + h * (3 * DHEAD);
    float acc = 0.f;
#pragma unroll 8
    for (int i = 0; i < DHEAD; ++i) acc += kp[i] * qp[i];
    acc *= 0.14433756729740643f; // 1/sqrt(48)
    score[t] = acc;
    atomicMax(&mEnc[d * NHEAD + h], gpfn_encodeF(acc));
}

__global__ void gpfn_edge_exp_kernel(const int* __restrict__ dst,
                                     float* __restrict__ score,
                                     const unsigned* __restrict__ mEnc,
                                     float* __restrict__ z) {
    int t = blockIdx.x * blockDim.x + threadIdx.x;
    if (t >= NEDGES * NHEAD) return;
    const int e = t >> 2, h = t & 3;
    const int d = dst[e];
    const float m = gpfn_decodeF(mEnc[d * NHEAD + h]);
    const float ex = expf(score[t] - m);
    score[t] = ex;
    atomicAdd(&z[d * NHEAD + h], ex);
}

__global__ void gpfn_edge_agg_kernel(const int* __restrict__ src,
                                     const int* __restrict__ dst,
                                     const float* __restrict__ score,
                                     const float* __restrict__ z,
                                     const float* __restrict__ qkv,
                                     float* __restrict__ agg) {
    int t = blockIdx.x * blockDim.x + threadIdx.x;
    if (t >= NEDGES * NHEAD) return;
    const int e = t >> 2, h = t & 3;
    const int s = src[e], d = dst[e];
    const float p = score[t] / z[d * NHEAD + h];
    const float* vp = qkv + (size_t)s * (3 * DMODEL) + h * (3 * DHEAD) + 2 * DHEAD;
    float* ap = agg + (size_t)d * DMODEL + h * DHEAD;
#pragma unroll 8
    for (int i = 0; i < DHEAD; ++i) atomicAdd(&ap[i], p * vp[i]);
}

// ---------------- host launcher ----------------
extern "C" void kernel_launch(void* const* d_in, const int* in_sizes, int n_in,
                              void* d_out, int out_size, void* d_ws, size_t ws_size,
                              hipStream_t stream) {
    const float*         x     = (const float*)d_in[0];
    const unsigned char* mask  = (const unsigned char*)d_in[1];
    const int*           src   = (const int*)d_in[2];
    const int*           dst   = (const int*)d_in[3];
    const float*         ln1_g = (const float*)d_in[4];
    const float*         ln1_b = (const float*)d_in[5];
    const float*         Wqkv  = (const float*)d_in[6];
    const float*         bqkv  = (const float*)d_in[7];
    const float*         Wout  = (const float*)d_in[8];
    const float*         bout  = (const float*)d_in[9];
    const float*         ln2_g = (const float*)d_in[10];
    const float*         ln2_b = (const float*)d_in[11];
    const float*         W1    = (const float*)d_in[12];
    const float*         b1    = (const float*)d_in[13];
    const float*         W2    = (const float*)d_in[14];
    const float*         b2    = (const float*)d_in[15];
    float* out = (float*)d_out;

    // workspace layout
    size_t off = 0;
    auto carve = [&](size_t bytes) -> void* {
        void* p = (char*)d_ws + off;
        off += (bytes + 255) & ~(size_t)255;
        return p;
    };
    int*       pos   = (int*)carve((size_t)NNODES * 4);
    int*       invp  = (int*)carve((size_t)NNODES * 4);
    float*     hperm = (float*)carve((size_t)NNODES * DMODEL * 4);
    _Float16*  hn1   = (_Float16*)carve((size_t)NNODES * DMODEL * 2);
    float*     qkv   = (float*)carve((size_t)NNODES * 3 * DMODEL * 4);
    float*     score = (float*)carve((size_t)NEDGES * NHEAD * 4);
    unsigned*  mEnc  = (unsigned*)carve((size_t)NNODES * NHEAD * 4);
    float*     zsum  = (float*)carve((size_t)NNODES * NHEAD * 4);
    float*     agg   = (float*)carve((size_t)NNODES * DMODEL * 4);
    float*     h2    = (float*)carve((size_t)NNODES * DMODEL * 4);
    float*     h3    = (float*)carve((size_t)NNODES * DMODEL * 4);
    float*     hn2   = (float*)carve((size_t)NNODES * DMODEL * 4);
    float*     act1  = (float*)carve((size_t)NNODES * 2 * DMODEL * 4);
    _Float16*  WqkvT = (_Float16*)carve((size_t)DMODEL * 3 * DMODEL * 2);
    _Float16*  WoutT = (_Float16*)carve((size_t)DMODEL * DMODEL * 2);
    _Float16*  W1T   = (_Float16*)carve((size_t)DMODEL * 2 * DMODEL * 2);
    _Float16*  W2T   = (_Float16*)carve((size_t)2 * DMODEL * DMODEL * 2);

    // init accumulators (must be re-done every call)
    hipMemsetAsync(agg,  0, (size_t)NNODES * DMODEL * 4, stream);
    hipMemsetAsync(zsum, 0, (size_t)NNODES * NHEAD * 4, stream);
    gpfn_init_m_kernel<<<(NNODES * NHEAD + 255) / 256, 256, 0, stream>>>(mEnc, NNODES * NHEAD);

    // weight convert + transpose
    gpfn_convert_wt_kernel<<<(DMODEL * 3 * DMODEL + 255) / 256, 256, 0, stream>>>(Wqkv, WqkvT, DMODEL, 3 * DMODEL);
    gpfn_convert_wt_kernel<<<(DMODEL * DMODEL + 255) / 256, 256, 0, stream>>>(Wout, WoutT, DMODEL, DMODEL);
    gpfn_convert_wt_kernel<<<(DMODEL * 2 * DMODEL + 255) / 256, 256, 0, stream>>>(W1, W1T, DMODEL, 2 * DMODEL);
    gpfn_convert_wt_kernel<<<(2 * DMODEL * DMODEL + 255) / 256, 256, 0, stream>>>(W2, W2T, 2 * DMODEL, DMODEL);

    // permutation
    gpfn_perm_kernel<<<1, 1024, 0, stream>>>(mask, pos, invp);

    // gather(order) + LN1 -> hperm (f32 residual), hn1 (f16 GEMM input)
    gpfn_gather_ln_kernel<_Float16><<<NNODES / 8, 256, 0, stream>>>(x, pos, ln1_g, ln1_b, hperm, hn1);

    // QKV GEMM: [N,192] @ [192,576]
    {
        const int waves = (NNODES / 16) * (3 * DMODEL / 64);
        gpfn_wmma_gemm_kernel<_Float16, DMODEL, 3 * DMODEL, 0>
            <<<(waves + 7) / 8, 256, 0, stream>>>(hn1, WqkvT, bqkv, nullptr, qkv, NNODES);
    }

    // edge attention
    const int et = NEDGES * NHEAD;
    gpfn_edge_score_kernel<<<(et + 255) / 256, 256, 0, stream>>>(src, dst, qkv, score, mEnc);
    gpfn_edge_exp_kernel<<<(et + 255) / 256, 256, 0, stream>>>(dst, score, mEnc, zsum);
    gpfn_edge_agg_kernel<<<(et + 255) / 256, 256, 0, stream>>>(src, dst, score, zsum, qkv, agg);

    // output projection + residual: h2 = hperm + agg @ Wout + bout
    {
        const int waves = (NNODES / 16) * (DMODEL / 64);
        gpfn_wmma_gemm_kernel<float, DMODEL, DMODEL, 1>
            <<<(waves + 7) / 8, 256, 0, stream>>>(agg, WoutT, bout, hperm, h2, NNODES);
    }

    // unpermute (inv_order) + LN2 -> h3 (f32 residual), hn2 (f32 GEMM input)
    gpfn_gather_ln_kernel<float><<<NNODES / 8, 256, 0, stream>>>(h2, invp, ln2_g, ln2_b, h3, hn2);

    // FFN1: act1 = gelu(hn2 @ W1 + b1)
    {
        const int waves = (NNODES / 16) * (2 * DMODEL / 64);
        gpfn_wmma_gemm_kernel<float, DMODEL, 2 * DMODEL, 2>
            <<<(waves + 7) / 8, 256, 0, stream>>>(hn2, W1T, b1, nullptr, act1, NNODES);
    }

    // FFN2 + residual -> output: out = h3 + act1 @ W2 + b2
    {
        const int waves = (NNODES / 16) * (DMODEL / 64);
        gpfn_wmma_gemm_kernel<float, 2 * DMODEL, DMODEL, 1>
            <<<(waves + 7) / 8, 256, 0, stream>>>(act1, W2T, b2, h3, out, NNODES);
    }
}
